// InjectedEncoder_54417235640880
// MI455X (gfx1250) — matrix operand order repaired and checked by reading, not compiled, not run on hardware
//
#include <hip/hip_runtime.h>
#include <cmath>

typedef _Float16 h16;
typedef __attribute__((ext_vector_type(16))) _Float16 v16h;
typedef __attribute__((ext_vector_type(8)))  float    v8f;

static constexpr int B_   = 4;
static constexpr int IMG_ = 128;
static constexpr int NPIX = IMG_ * IMG_;        // 16384
static constexpr int NF_  = 64;
static constexpr int LAT_ = 512;
static constexpr int S3D  = 26 * 26 * 26;       // 17576
static constexpr int NTILES = 275;              // ceil(17576/64)
static constexpr int FIN_ = 448;
static constexpr int FOUT_ = 128;
static constexpr int PTOT_ = 98688;
static constexpr int OFF_W1 = 0, OFF_B1 = 28672, OFF_W2 = 28736, OFF_B2 = 32832;
static constexpr int OFF_W3 = 32896, OFF_B3 = 41088, OFF_WSH = 41216, OFF_BSH = 98560;
static constexpr int OUTC_ = 193;
static constexpr int STATE_ = 32832;            // 64 + 8192 + 8192 + 16384

__device__ __forceinline__ float lrelu_f(float x) { return fmaxf(x, 0.2f * x); }

// ---- WMMA fragment helpers (CDNA5 v_wmma_f32_16x16x32_f16) ----------------
// A 16x32 f16 (row-major source, stride ldw): lane<16 -> row lane%16,
//   halves 0..7 = K[kb..kb+7], halves 8..15 = K[kb+16..kb+23]; lanes>=16: kb+=8.
__device__ __forceinline__ v16h ld_a_frag(const h16* W, int ldw, int row, int kb) {
  union { uint4 u[2]; v16h v; } t;
  const h16* p = W + (size_t)row * ldw + kb;
  t.u[0] = *(const uint4*)(p);
  t.u[1] = *(const uint4*)(p + 16);
  return t.v;
}
// B 32x16 f16 from LDS tile stored [col][k] (stride ldp): lane<16 -> col lane%16,
//   halves 0..7 = K[kb..kb+7], halves 8..15 = K[kb+16..kb+23]; lanes>=16: kb+=8.
__device__ __forceinline__ v16h ld_b_frag(const h16* P, int ldp, int col, int kb) {
  union { uint4 u[2]; v16h v; } t;
  const h16* p = P + (size_t)col * ldp + kb;
  t.u[0] = *(const uint4*)(p);
  t.u[1] = *(const uint4*)(p + 16);
  return t.v;
}
__device__ __forceinline__ v8f wmma16(v16h a, v16h b, v8f c) {
  return __builtin_amdgcn_wmma_f32_16x16x32_f16(false, a, false, b, (short)0, c, false, false);
}

// ---- hypernetwork: dp = inj_lat @ hyper_w + hyper_b; also f16 copy --------
__global__ void __launch_bounds__(256) k_hyper(const float* __restrict__ lat,
                                               const float* __restrict__ hw,
                                               const float* __restrict__ hb,
                                               float* __restrict__ dp,
                                               h16* __restrict__ dp16) {
  __shared__ float xl[LAT_];
  int b = blockIdx.y;
  for (int i = threadIdx.x; i < LAT_; i += 256) xl[i] = lat[b * LAT_ + i];
  __syncthreads();
  int col = blockIdx.x * 256 + threadIdx.x;
  if (col >= PTOT_) return;
  float acc = hb[col];
  for (int k = 0; k < LAT_; ++k) acc = fmaf(xl[k], hw[(size_t)k * PTOT_ + col], acc);
  dp[(size_t)b * PTOT_ + col] = acc;
  dp16[(size_t)b * PTOT_ + col] = (h16)acc;
}

// ---- input 1x1 conv + rb2d -> carry cube (first NPIX entries) -------------
__global__ void __launch_bounds__(256) k_pre(const float* __restrict__ x,
                                             const float* __restrict__ in_w, const float* __restrict__ in_b,
                                             const float* __restrict__ w1, const float* __restrict__ b1,
                                             const float* __restrict__ w2, const float* __restrict__ b2,
                                             float* __restrict__ carry) {
  __shared__ float sw1[NF_ * NF_], sw2[NF_ * NF_], siw[NF_ * 3];
  __shared__ float sib[NF_], sb1[NF_], sb2[NF_];
  int b = blockIdx.y;
  for (int i = threadIdx.x; i < NF_ * NF_; i += 256) { sw1[i] = w1[i]; sw2[i] = w2[i]; }
  for (int i = threadIdx.x; i < NF_ * 3; i += 256) siw[i] = in_w[i];
  for (int i = threadIdx.x; i < NF_; i += 256) { sib[i] = in_b[i]; sb1[i] = b1[i]; sb2[i] = b2[i]; }
  __syncthreads();
  int p = blockIdx.x * 256 + threadIdx.x;
  float x0 = x[((size_t)b * 3 + 0) * NPIX + p];
  float x1 = x[((size_t)b * 3 + 1) * NPIX + p];
  float x2 = x[((size_t)b * 3 + 2) * NPIX + p];
  float h[NF_], t[NF_];
#pragma unroll
  for (int o = 0; o < NF_; ++o)
    h[o] = sib[o] + siw[o * 3] * x0 + siw[o * 3 + 1] * x1 + siw[o * 3 + 2] * x2;
#pragma unroll
  for (int o = 0; o < NF_; ++o) {
    float a = sb1[o];
#pragma unroll
    for (int c = 0; c < NF_; ++c) a = fmaf(sw1[o * 64 + c], h[c], a);
    t[o] = lrelu_f(a);
  }
#pragma unroll
  for (int o = 0; o < NF_; ++o) {
    float a = sb2[o];
#pragma unroll
    for (int c = 0; c < NF_; ++c) a = fmaf(sw2[o * 64 + c], t[c], a);
    carry[((size_t)b * NF_ + o) * S3D + p] = h[o] + a;
  }
}

__global__ void k_padtail(float* __restrict__ carry) {
  int i = blockIdx.x * 256 + threadIdx.x;
  const int tail = S3D - NPIX;
  if (i >= B_ * NF_ * tail) return;
  int off = i % tail, bc = i / tail;
  carry[(size_t)bc * S3D + NPIX + off] = 0.f;
}

__global__ void k_zero(float* __restrict__ p, int n) {
  int i = blockIdx.x * 256 + threadIdx.x;
  if (i < n) p[i] = 0.f;
}

// ---- sobel (depthwise 3D, sparse taps), z-slab split, per-slab stats ------
// grid: (channel 0..63, slab 0..3, batch). Slab z-extents {7,7,6,6}.
__global__ void __launch_bounds__(256) k_sobel(const float* __restrict__ carry,
                                               h16* __restrict__ praw,
                                               float* __restrict__ sumP,
                                               float* __restrict__ sqP) {
  __shared__ float cube[11 * 900];   // up to (7+4) z-planes with halo 2 (39.6 KB)
  int c = blockIdx.x, slab = blockIdx.y, b = blockIdx.z;
  int z0 = (slab < 2) ? slab * 7 : 14 + (slab - 2) * 6;
  int nz = (slab < 2) ? 7 : 6;
  const float* src = carry + ((size_t)b * NF_ + c) * S3D;
  for (int i = threadIdx.x; i < (nz + 4) * 900; i += 256) {
    int zp = i / 900, r = i % 900, y = r / 30, xx = r % 30;
    int z = z0 + zp - 2;
    float v = 0.f;
    if (z >= 0 && z < 26 && y >= 2 && y < 28 && xx >= 2 && xx < 28)
      v = src[z * 676 + (y - 2) * 26 + (xx - 2)];
    cube[i] = v;
  }
  __syncthreads();
  constexpr float g3[3] = {-1.f, 0.f, 1.f};
  constexpr float s3[3] = {0.f, 1.f, 0.f};
  constexpr float g5[5] = {-1.f, -0.70710678f, 0.f, 0.70710678f, 1.f};
  constexpr float s5[5] = {0.f, 0.70710678f, 1.f, 0.70710678f, 0.f};
  float sum[7] = {0, 0, 0, 0, 0, 0, 0}, sq[7] = {0, 0, 0, 0, 0, 0, 0};
  h16* outp = praw + (size_t)b * FIN_ * S3D;
  int nvox = nz * 676;
  for (int v = threadIdx.x; v < nvox; v += 256) {
    int zl = v / 676, r = v % 676, y = r / 26, xx = r % 26;
    int base = (zl + 2) * 900 + (y + 2) * 30 + (xx + 2);
    int gs = (z0 + zl) * 676 + r;   // global voxel index
    float o[7];
    o[0] = cube[base];
    float ax = 0.f, ay = 0.f, az = 0.f;
#pragma unroll
    for (int jz = 0; jz < 3; ++jz)
#pragma unroll
      for (int jy = 0; jy < 3; ++jy)
#pragma unroll
        for (int jx = 0; jx < 3; ++jx) {
          float wx = g3[jz] * s3[jy] * s3[jx];
          float wy = s3[jz] * g3[jy] * s3[jx];
          float wz = s3[jz] * s3[jy] * g3[jx];
          if (wx == 0.f && wy == 0.f && wz == 0.f) continue;   // folds at compile time
          float val = cube[base + (jz - 1) * 900 + (jy - 1) * 30 + (jx - 1)];
          ax = fmaf(wx, val, ax); ay = fmaf(wy, val, ay); az = fmaf(wz, val, az);
        }
    o[1] = ax; o[2] = ay; o[3] = az;
    ax = ay = az = 0.f;
#pragma unroll
    for (int jz = 0; jz < 5; ++jz)
#pragma unroll
      for (int jy = 0; jy < 5; ++jy)
#pragma unroll
        for (int jx = 0; jx < 5; ++jx) {
          float wx = g5[jz] * s5[jy] * s5[jx];
          float wy = s5[jz] * g5[jy] * s5[jx];
          float wz = s5[jz] * s5[jy] * g5[jx];
          if (wx == 0.f && wy == 0.f && wz == 0.f) continue;
          float val = cube[base + (jz - 2) * 900 + (jy - 2) * 30 + (jx - 2)];
          ax = fmaf(wx, val, ax); ay = fmaf(wy, val, ay); az = fmaf(wz, val, az);
        }
    o[4] = ax; o[5] = ay; o[6] = az;
#pragma unroll
    for (int q = 0; q < 7; ++q) {
      sum[q] += o[q];
      sq[q] = fmaf(o[q], o[q], sq[q]);
      int chq = (q == 0) ? c : (q < 4 ? 64 + c * 3 + (q - 1) : 256 + c * 3 + (q - 4));
      outp[(size_t)chq * S3D + gs] = (h16)o[q];
    }
  }
  __syncthreads();
  float* red = cube;   // reuse LDS for reductions (per-slab partials, no atomics)
#pragma unroll
  for (int q = 0; q < 7; ++q) {
    red[threadIdx.x] = sum[q];
    red[256 + threadIdx.x] = sq[q];
    __syncthreads();
    for (int s = 128; s > 0; s >>= 1) {
      if ((int)threadIdx.x < s) {
        red[threadIdx.x] += red[threadIdx.x + s];
        red[256 + threadIdx.x] += red[256 + threadIdx.x + s];
      }
      __syncthreads();
    }
    if (threadIdx.x == 0) {
      int chq = (q == 0) ? c : (q < 4 ? 64 + c * 3 + (q - 1) : 256 + c * 3 + (q - 4));
      sumP[((size_t)b * FIN_ + chq) * 4 + slab] = red[0];
      sqP[((size_t)b * FIN_ + chq) * 4 + slab] = red[256];
    }
    __syncthreads();
  }
}

// ---- fused per-step GEMM chain (WMMA), normalize-on-load, gated update ----
__global__ void __launch_bounds__(256) k_step_gemm(const h16* __restrict__ praw,
                                                   const float* __restrict__ sumP,
                                                   const float* __restrict__ sqP,
                                                   const h16* __restrict__ dp16,
                                                   const float* __restrict__ dp,
                                                   const float* __restrict__ leak_p,
                                                   float* __restrict__ carry) {
  __shared__ __align__(16) h16 Pt[64 * FIN_];   // p tile, [col][c]        (57344 B)
  __shared__ __align__(16) h16 H1[64 * 64];     // hh1,   [col][c]         ( 8192 B)
  __shared__ __align__(16) h16 H2[64 * 64];     // hh2,   [col][c]         ( 8192 B)
  __shared__ float Yt[FOUT_ * 64];              // y,     [o][col]         (32768 B)
  __shared__ float ms[FIN_], rs[FIN_];          // inorm mean / rstd       ( 3584 B)
  int b = blockIdx.y, col0 = blockIdx.x * 64;
  int tid = threadIdx.x;
  // per-channel instance-norm stats from deterministic per-slab partials
  for (int c = tid; c < FIN_; c += 256) {
    const float* sp = sumP + ((size_t)b * FIN_ + c) * 4;
    const float* qp = sqP + ((size_t)b * FIN_ + c) * 4;
    float s = (sp[0] + sp[1]) + (sp[2] + sp[3]);
    float q = (qp[0] + qp[1]) + (qp[2] + qp[3]);
    float m = s * (1.f / S3D);
    float var = q * (1.f / S3D) - m * m;
    ms[c] = m;
    rs[c] = rsqrtf(fmaxf(var, 0.f) + 1e-5f);
  }
  __syncthreads();
  // normalize + transpose praw[c][s] -> Pt[col][c] (f16), fused (no p16 pass)
  const h16* src = praw + (size_t)b * FIN_ * S3D;
  if (col0 + 64 <= S3D) {
    for (int i = tid; i < FIN_ * 16; i += 256) {  // 4 halves (8 B) per iteration
      int c = i >> 4, g = i & 15;
      union { uint2 u; h16 hv[4]; } tt;
      tt.u = *(const uint2*)(src + (size_t)c * S3D + col0 + g * 4);
      float m = ms[c], r = rs[c];
#pragma unroll
      for (int e = 0; e < 4; ++e)
        Pt[(g * 4 + e) * FIN_ + c] = (h16)(((float)tt.hv[e] - m) * r);
    }
  } else {
    for (int i = tid; i < FIN_ * 64; i += 256) {
      int c = i >> 6, sc = i & 63;
      int s = col0 + sc;
      float v = 0.f;
      if (s < S3D) v = ((float)src[(size_t)c * S3D + s] - ms[c]) * rs[c];
      Pt[sc * FIN_ + c] = (h16)v;
    }
  }
  __syncthreads();
  const h16* W1 = dp16 + (size_t)b * PTOT_ + OFF_W1;
  const h16* W2 = dp16 + (size_t)b * PTOT_ + OFF_W2;
  const h16* W3 = dp16 + (size_t)b * PTOT_ + OFF_W3;
  const h16* WS = dp16 + (size_t)b * PTOT_ + OFF_WSH;
  const float* dB = dp + (size_t)b * PTOT_;
  int w = tid >> 5, lane = tid & 31;
  int lh = lane >> 4, lm = lane & 15;
  int mi = w >> 2, ni = w & 3;                 // wave handles row tiles mi, mi+2 (cols ni)
  int col = ni * 16 + lm;
  // stage 1: hh1 = lrelu(W1 @ p + b1), 64x64
  {
    v8f a0 = {}, a1 = {};
#pragma unroll
    for (int r = 0; r < 8; ++r) {
      a0[r] = dB[OFF_B1 + mi * 16 + 8 * lh + r];
      a1[r] = dB[OFF_B1 + (mi + 2) * 16 + 8 * lh + r];
    }
    for (int k = 0; k < FIN_ / 32; ++k) {
      v16h bf = ld_b_frag(Pt, FIN_, col, k * 32 + lh * 8);
      a0 = wmma16(ld_a_frag(W1, FIN_, mi * 16 + lm, k * 32 + lh * 8), bf, a0);
      a1 = wmma16(ld_a_frag(W1, FIN_, (mi + 2) * 16 + lm, k * 32 + lh * 8), bf, a1);
    }
#pragma unroll
    for (int r = 0; r < 8; ++r) {
      H1[col * 64 + mi * 16 + 8 * lh + r] = (h16)lrelu_f(a0[r]);
      H1[col * 64 + (mi + 2) * 16 + 8 * lh + r] = (h16)lrelu_f(a1[r]);
    }
  }
  __syncthreads();
  // stage 2: hh2 = lrelu(W2 @ hh1 + b2), 64x64
  {
    v8f a0 = {}, a1 = {};
#pragma unroll
    for (int r = 0; r < 8; ++r) {
      a0[r] = dB[OFF_B2 + mi * 16 + 8 * lh + r];
      a1[r] = dB[OFF_B2 + (mi + 2) * 16 + 8 * lh + r];
    }
#pragma unroll
    for (int k = 0; k < 2; ++k) {
      v16h bf = ld_b_frag(H1, 64, col, k * 32 + lh * 8);
      a0 = wmma16(ld_a_frag(W2, 64, mi * 16 + lm, k * 32 + lh * 8), bf, a0);
      a1 = wmma16(ld_a_frag(W2, 64, (mi + 2) * 16 + lm, k * 32 + lh * 8), bf, a1);
    }
#pragma unroll
    for (int r = 0; r < 8; ++r) {
      H2[col * 64 + mi * 16 + 8 * lh + r] = (h16)lrelu_f(a0[r]);
      H2[col * 64 + (mi + 2) * 16 + 8 * lh + r] = (h16)lrelu_f(a1[r]);
    }
  }
  __syncthreads();
  // stage 3: y = W3 @ hh2 + Wsh @ p + b3 + bsh, 128x64 (4 row tiles per wave)
  {
    v8f acc[4];
#pragma unroll
    for (int j = 0; j < 4; ++j)
#pragma unroll
      for (int r = 0; r < 8; ++r) {
        int o = (mi + 2 * j) * 16 + 8 * lh + r;
        acc[j][r] = dB[OFF_B3 + o] + dB[OFF_BSH + o];
      }
    for (int k = 0; k < FIN_ / 32; ++k) {
      v16h bf = ld_b_frag(Pt, FIN_, col, k * 32 + lh * 8);
#pragma unroll
      for (int j = 0; j < 4; ++j)
        acc[j] = wmma16(ld_a_frag(WS, FIN_, (mi + 2 * j) * 16 + lm, k * 32 + lh * 8), bf, acc[j]);
    }
#pragma unroll
    for (int k = 0; k < 2; ++k) {
      v16h bf = ld_b_frag(H2, 64, col, k * 32 + lh * 8);
#pragma unroll
      for (int j = 0; j < 4; ++j)
        acc[j] = wmma16(ld_a_frag(W3, 64, (mi + 2 * j) * 16 + lm, k * 32 + lh * 8), bf, acc[j]);
    }
#pragma unroll
    for (int j = 0; j < 4; ++j)
#pragma unroll
      for (int r = 0; r < 8; ++r)
        Yt[((mi + 2 * j) * 16 + 8 * lh + r) * 64 + col] = acc[j][r];
  }
  __syncthreads();
  // stage 4: carry += clip(leak) * val * sigmoid(gate)
  float leak = fminf(fmaxf(*leak_p, 1e-3f), 1000.f);
  for (int i = tid; i < 64 * 64; i += 256) {
    int c = i >> 6, sc = i & 63;
    int s = col0 + sc;
    if (s < S3D) {
      float val = Yt[c * 64 + sc];
      float gate = Yt[(c + 64) * 64 + sc];
      float sig = 1.f / (1.f + __expf(-gate));
      size_t idx = ((size_t)b * NF_ + c) * S3D + s;
      carry[idx] = fmaf(leak, val * sig, carry[idx]);
    }
  }
}

// ---- epilogue: rb2d + outc conv + pooled state vector ---------------------
__global__ void __launch_bounds__(128) k_post(const float* __restrict__ carry,
                                              const float* __restrict__ w1, const float* __restrict__ b1,
                                              const float* __restrict__ w2, const float* __restrict__ b2,
                                              const float* __restrict__ ow, const float* __restrict__ ob,
                                              float* __restrict__ state) {
  __shared__ float sw1[4096], sw2[4096], sow[OUTC_ * 64], sob[OUTC_], sb1[64], sb2[64];
  __shared__ float hb[128 * 65];   // padded to kill bank conflicts
  __shared__ float redw[4];
  int b = blockIdx.y, hrow = blockIdx.x;
  for (int i = threadIdx.x; i < 4096; i += 128) { sw1[i] = w1[i]; sw2[i] = w2[i]; }
  for (int i = threadIdx.x; i < OUTC_ * 64; i += 128) sow[i] = ow[i];
  for (int i = threadIdx.x; i < OUTC_; i += 128) sob[i] = ob[i];
  for (int i = threadIdx.x; i < 64; i += 128) { sb1[i] = b1[i]; sb2[i] = b2[i]; }
  __syncthreads();
  int wcol = threadIdx.x;
  size_t pix = (size_t)hrow * 128 + wcol;
  float h[64], t[64];
#pragma unroll
  for (int c = 0; c < 64; ++c) h[c] = carry[((size_t)b * NF_ + c) * S3D + pix];
#pragma unroll
  for (int o = 0; o < 64; ++o) {
    float a = sb1[o];
#pragma unroll
    for (int c = 0; c < 64; ++c) a = fmaf(sw1[o * 64 + c], h[c], a);
    t[o] = lrelu_f(a);
  }
#pragma unroll
  for (int o = 0; o < 64; ++o) {
    float a = sb2[o];
#pragma unroll
    for (int c = 0; c < 64; ++c) a = fmaf(sw2[o * 64 + c], t[c], a);
    hb[wcol * 65 + o] = h[o] + a;
  }
  __syncthreads();
  float* st = state + (size_t)b * STATE_;
  int lane = threadIdx.x & 31, wv = threadIdx.x >> 5;
  for (int o = 0; o < OUTC_; ++o) {
    float v = sob[o];
#pragma unroll
    for (int c = 0; c < 64; ++c) v = fmaf(sow[o * 64 + c], hb[wcol * 65 + c], v);
    if (o < 128) {               // block reduction over the 128 w-columns
      float r = v;
#pragma unroll
      for (int m = 16; m > 0; m >>= 1) r += __shfl_xor(r, m, 32);
      if (lane == 0) redw[wv] = r;
      __syncthreads();
      if (threadIdx.x == 0) {
        float s = redw[0] + redw[1] + redw[2] + redw[3];
        if (o < 64) atomicAdd(&st[o], s * (1.f / 16384.f));
        else st[64 + (o - 64) * 128 + hrow] = s * (1.f / 128.f);
      }
      __syncthreads();
    } else if (o < 192) {
      atomicAdd(&st[8256 + (o - 128) * 128 + wcol], v * (1.f / 128.f));
    } else {
      st[16448 + (size_t)hrow * 128 + wcol] = v;
    }
  }
}

// ---- generic MLP matvec: Y = act(X@W + bias [+ Xs@Ws] [+ skip]) -----------
__global__ void __launch_bounds__(256) k_mv(const float* __restrict__ X,
                                            const float* __restrict__ W,
                                            const float* __restrict__ bias,
                                            const float* __restrict__ Xs,
                                            const float* __restrict__ Ws,
                                            const float* __restrict__ skip,
                                            float* __restrict__ Y,
                                            int IN, int INs, int OUT, int act) {
  __shared__ float xc[256];
  int b = blockIdx.y;
  int o = blockIdx.x * 256 + threadIdx.x;
  float acc = (o < OUT) ? bias[o] : 0.f;
  const float* xb = X + (size_t)b * IN;
  for (int k0 = 0; k0 < IN; k0 += 256) {
    int n = min(256, IN - k0);
    __syncthreads();
    if ((int)threadIdx.x < n) xc[threadIdx.x] = xb[k0 + threadIdx.x];
    __syncthreads();
    if (o < OUT)
      for (int k = 0; k < n; ++k) acc = fmaf(xc[k], W[(size_t)(k0 + k) * OUT + o], acc);
  }
  if (Ws) {
    const float* xs = Xs + (size_t)b * INs;
    for (int k0 = 0; k0 < INs; k0 += 256) {
      int n = min(256, INs - k0);
      __syncthreads();
      if ((int)threadIdx.x < n) xc[threadIdx.x] = xs[k0 + threadIdx.x];
      __syncthreads();
      if (o < OUT)
        for (int k = 0; k < n; ++k) acc = fmaf(xc[k], Ws[(size_t)(k0 + k) * OUT + o], acc);
    }
  }
  if (o >= OUT) return;
  if (skip) acc += skip[(size_t)b * OUT + o];
  if (act) acc = lrelu_f(acc);
  Y[(size_t)b * OUT + o] = acc;
}

// ---------------------------------------------------------------------------
extern "C" void kernel_launch(void* const* d_in, const int* in_sizes, int n_in,
                              void* d_out, int out_size, void* d_ws, size_t ws_size,
                              hipStream_t stream) {
  const float* x       = (const float*)d_in[0];
  const float* inj_lat = (const float*)d_in[1];
  const float* in_w    = (const float*)d_in[2];
  const float* in_b    = (const float*)d_in[3];
  const float* rbin_w1 = (const float*)d_in[4];
  const float* rbin_b1 = (const float*)d_in[5];
  const float* rbin_w2 = (const float*)d_in[6];
  const float* rbin_b2 = (const float*)d_in[7];
  const float* leak    = (const float*)d_in[8];
  const float* hyper_w = (const float*)d_in[9];
  const float* hyper_b = (const float*)d_in[10];
  const float* rbout_w1= (const float*)d_in[11];
  const float* rbout_b1= (const float*)d_in[12];
  const float* rbout_w2= (const float*)d_in[13];
  const float* rbout_b2= (const float*)d_in[14];
  const float* outc_w  = (const float*)d_in[15];
  const float* outc_b  = (const float*)d_in[16];
  const float* l1a_w   = (const float*)d_in[17];
  const float* l1a_b   = (const float*)d_in[18];
  const float* l1b_w   = (const float*)d_in[19];
  const float* l1b_b   = (const float*)d_in[20];
  const float* l1s_w   = (const float*)d_in[21];
  const float* l2a_w   = (const float*)d_in[22];
  const float* l2a_b   = (const float*)d_in[23];
  const float* l2b_w   = (const float*)d_in[24];
  const float* l2b_b   = (const float*)d_in[25];
  const float* lat_w   = (const float*)d_in[26];
  const float* lat_b   = (const float*)d_in[27];
  (void)in_sizes; (void)n_in; (void)out_size; (void)ws_size;

  char* wp = (char*)d_ws;
  auto alloc = [&](size_t bytes) { void* r = (void*)wp; wp += (bytes + 255) & ~(size_t)255; return r; };
  float* carry = (float*)alloc((size_t)B_ * NF_ * S3D * 4);
  h16*   praw  = (h16*)  alloc((size_t)B_ * FIN_ * S3D * 2);
  float* sumP  = (float*)alloc((size_t)B_ * FIN_ * 4 * 4);
  float* sqP   = (float*)alloc((size_t)B_ * FIN_ * 4 * 4);
  float* dp    = (float*)alloc((size_t)B_ * PTOT_ * 4);
  h16*   dp16  = (h16*)  alloc((size_t)B_ * PTOT_ * 2);
  float* state = (float*)alloc((size_t)B_ * STATE_ * 4);
  float* t1    = (float*)alloc((size_t)B_ * 1024 * 4);
  float* z1    = (float*)alloc((size_t)B_ * 512 * 4);
  float* t2    = (float*)alloc((size_t)B_ * 512 * 4);
  float* z2    = (float*)alloc((size_t)B_ * 512 * 4);

  k_hyper<<<dim3((PTOT_ + 255) / 256, B_), 256, 0, stream>>>(inj_lat, hyper_w, hyper_b, dp, dp16);
  k_pre<<<dim3(NPIX / 256, B_), 256, 0, stream>>>(x, in_w, in_b, rbin_w1, rbin_b1, rbin_w2, rbin_b2, carry);
  k_padtail<<<(B_ * NF_ * (S3D - NPIX) + 255) / 256, 256, 0, stream>>>(carry);
  k_zero<<<(B_ * STATE_ + 255) / 256, 256, 0, stream>>>(state, B_ * STATE_);

  for (int step = 0; step < 4; ++step) {
    k_sobel<<<dim3(NF_, 4, B_), 256, 0, stream>>>(carry, praw, sumP, sqP);
    k_step_gemm<<<dim3(NTILES, B_), 256, 0, stream>>>(praw, sumP, sqP, dp16, dp, leak, carry);
  }

  k_post<<<dim3(IMG_, B_), 128, 0, stream>>>(carry, rbout_w1, rbout_b1, rbout_w2, rbout_b2,
                                             outc_w, outc_b, state);
  k_mv<<<dim3(4, B_), 256, 0, stream>>>(state, l1a_w, l1a_b, nullptr, nullptr, nullptr, t1, STATE_, 0, 1024, 1);
  k_mv<<<dim3(2, B_), 256, 0, stream>>>(t1, l1b_w, l1b_b, state, l1s_w, nullptr, z1, 1024, STATE_, 512, 0);
  k_mv<<<dim3(2, B_), 256, 0, stream>>>(z1, l2a_w, l2a_b, nullptr, nullptr, nullptr, t2, 512, 0, 512, 1);
  k_mv<<<dim3(2, B_), 256, 0, stream>>>(t2, l2b_w, l2b_b, nullptr, nullptr, z1, z2, 512, 0, 512, 0);
  k_mv<<<dim3(2, B_), 256, 0, stream>>>(z2, lat_w, lat_b, nullptr, nullptr, nullptr, (float*)d_out, 512, 0, 512, 0);
}
